// MultiHeadAttention_19610820674016
// MI455X (gfx1250) — matrix operand order repaired and checked by reading, not compiled, not run on hardware
//
#include <hip/hip_runtime.h>
#include <hip/hip_bf16.h>

typedef _Float16 f16;
typedef __attribute__((ext_vector_type(16))) _Float16 v16h;
typedef __attribute__((ext_vector_type(8)))  _Float16 v8h;
typedef __attribute__((ext_vector_type(8)))  float    v8f;

constexpr int BQ = 4;      // batch
constexpr int TQ = 2048;   // seq len
constexpr int CQ = 1024;   // channels
constexpr int HQ = 16;     // heads
constexpr int DQ = 64;     // head dim
constexpr long MTOT = (long)BQ * TQ;   // 8192 rows
constexpr int LDP = 40;    // padded LDS row pitch in halfs (80 B, 16B-aligned)

// ---------------------------------------------------------------------------
// WMMA fragment helpers (CDNA5 ISA 7.12.2, wave32)
// A 16x32 f16: lane&15 = M row; half = lane>>4.
//   elems 0..15 = K[16*half .. +15]
// B 32x16 f16 (N-major, K-contiguous storage): lane&15 = N col.
//   elems 0..15 = K[16*half .. +15]
// C/D 16x16 f32: lane&15 = N; VGPR j = row M = 8*half + j.
// ---------------------------------------------------------------------------
__device__ __forceinline__ v16h make_frag(v8h lo, v8h hi) {
  v16h r;
#pragma unroll
  for (int i = 0; i < 8; ++i) { r[i] = lo[i]; r[8 + i] = hi[i]; }
  return r;
}

__device__ __forceinline__ v16h frag_a(const f16* p, int half) {
  v8h lo = *(const v8h*)(p + 16 * half);
  v8h hi = *(const v8h*)(p + 16 * half + 8);
  return make_frag(lo, hi);
}

__device__ __forceinline__ v16h frag_b(const f16* p, int half) {
  v8h lo = *(const v8h*)(p + 16 * half);
  v8h hi = *(const v8h*)(p + 16 * half + 8);
  return make_frag(lo, hi);
}

__device__ __forceinline__ v8f wmma_f16(v16h a, v16h b, v8f c) {
  return __builtin_amdgcn_wmma_f32_16x16x32_f16(false, a, false, b,
                                                (short)0, c, false, false);
}

// Async DMA: one 16-byte chunk global -> LDS (ASYNCcnt-tracked).
__device__ __forceinline__ void async_b128(unsigned lds_off, const f16* g) {
  asm volatile("global_load_async_to_lds_b128 %0, %1, off"
               :: "v"(lds_off), "v"(g) : "memory");
}

// ---------------------------------------------------------------------------
// Conversion / transpose kernels
// ---------------------------------------------------------------------------
__global__ void k_cvt_f16(const float* __restrict__ in, f16* __restrict__ out,
                          long n) {
  long i = (long)blockIdx.x * blockDim.x + threadIdx.x;
  const long stride = (long)gridDim.x * blockDim.x;
  for (; i < n; i += stride) out[i] = (f16)in[i];
}

// W[k][n] (row-major, applied as x @ W) -> Wt[n][k] f16 (N-major, K-contig).
__global__ void k_transpose_w(const float* __restrict__ W,
                              f16* __restrict__ Wt) {
  long idx = (long)blockIdx.x * blockDim.x + threadIdx.x;
  if (idx >= (long)CQ * CQ) return;
  const int k = (int)(idx % CQ);
  const int n = (int)(idx / CQ);
  Wt[idx] = (f16)W[(long)k * CQ + n];
}

// ---------------------------------------------------------------------------
// WMMA GEMM, LDS-staged with async DMA double buffering.
// out(M,N) = oscale * A(M,K) @ Bt(N,K)^T  [K = 1024]
// Block = 256 thr = 8 waves (4 M x 2 N); block tile 128x128; wave tile 32x64.
// MODE 0: f16 scatter to (B,H,T,D)   MODE 1: f16 scatter to (B,H,D,T)
// MODE 2: f32 row-major + bias
// ---------------------------------------------------------------------------
template <int MODE>
__global__ __launch_bounds__(256)
void k_gemm(const f16* __restrict__ A, const f16* __restrict__ Bt,
            f16* __restrict__ o16, float* __restrict__ o32,
            const float* __restrict__ bias, float oscale) {
  __shared__ f16 shA[2][128][LDP];
  __shared__ f16 shB[2][128][LDP];

  const int tid  = threadIdx.x;
  const int lane = tid & 31;
  const int wave = tid >> 5;
  const int mn   = lane & 15;
  const int half = lane >> 4;
  const int rowB = blockIdx.x * 128;
  const int colB = blockIdx.y * 128;
  const int wrow = (wave & 3) * 32;
  const int wcol = (wave >> 2) * 64;

  // Staging map: 128 rows x 4 16B-chunks = 512 chunks; 2 A + 2 B per thread.
  const int r0 = tid >> 1;
  const int c0 = (tid & 1) * 2;  // handles chunks c0, c0+1 (8 halfs each)
  const f16* gA = A + (long)(rowB + r0) * CQ + c0 * 8;
  const f16* gB = Bt + (long)(colB + r0) * CQ + c0 * 8;
  const unsigned lA[2] = {(unsigned)(size_t)&shA[0][r0][c0 * 8],
                          (unsigned)(size_t)&shA[1][r0][c0 * 8]};
  const unsigned lB[2] = {(unsigned)(size_t)&shB[0][r0][c0 * 8],
                          (unsigned)(size_t)&shB[1][r0][c0 * 8]};

  v8f acc[2][4] = {};

  // prologue prefetch of K-step 0
  async_b128(lA[0], gA);
  async_b128(lA[0] + 16, gA + 8);
  async_b128(lB[0], gB);
  async_b128(lB[0] + 16, gB + 8);

  const int NK = CQ / 32;
  for (int ks = 0; ks < NK; ++ks) {
    const int buf = ks & 1;
    if (ks + 1 < NK) {  // prefetch next K step into the other buffer
      const long ko = (long)(ks + 1) * 32;
      async_b128(lA[buf ^ 1], gA + ko);
      async_b128(lA[buf ^ 1] + 16, gA + ko + 8);
      async_b128(lB[buf ^ 1], gB + ko);
      async_b128(lB[buf ^ 1] + 16, gB + ko + 8);
      asm volatile("s_wait_asynccnt 0x4" ::: "memory");  // current buf done
    } else {
      asm volatile("s_wait_asynccnt 0x0" ::: "memory");
    }
    __syncthreads();  // all waves' DMA into buf complete

    v16h af[2], bf[4];
#pragma unroll
    for (int mi = 0; mi < 2; ++mi)
      af[mi] = frag_a(&shA[buf][wrow + mi * 16 + mn][0], half);
#pragma unroll
    for (int ni = 0; ni < 4; ++ni)
      bf[ni] = frag_b(&shB[buf][wcol + ni * 16 + mn][0], half);
#pragma unroll
    for (int mi = 0; mi < 2; ++mi)
#pragma unroll
      for (int ni = 0; ni < 4; ++ni)
        acc[mi][ni] = wmma_f16(af[mi], bf[ni], acc[mi][ni]);

    __syncthreads();  // done reading buf; next iter may overwrite it
  }

#pragma unroll
  for (int mi = 0; mi < 2; ++mi)
#pragma unroll
    for (int ni = 0; ni < 4; ++ni)
#pragma unroll
      for (int j = 0; j < 8; ++j) {
        const int r = rowB + wrow + mi * 16 + 8 * half + j;
        const int c = colB + wcol + ni * 16 + mn;
        const float v = acc[mi][ni][j] * oscale;
        if (MODE == 0) {
          const int b = r / TQ, t = r % TQ;
          const int h = c / DQ, d = c % DQ;
          o16[(((long)(b * HQ + h)) * TQ + t) * DQ + d] = (f16)v;
        } else if (MODE == 1) {
          const int b = r / TQ, t = r % TQ;
          const int h = c / DQ, d = c % DQ;
          o16[(((long)(b * HQ + h)) * DQ + d) * TQ + t] = (f16)v;
        } else {
          o32[(long)r * CQ + c] = v + bias[c];
        }
      }
}

// ---------------------------------------------------------------------------
// Flash attention, fully transposed formulation.
//   S^T = K Q^T      : lane = query, keys in VGPRs (in-lane row stats)
//   ctx^T = V^T P^T  : lane = query, d in VGPRs (in-lane corr/normalize,
//                      contiguous b128 ctx stores)
// P^T B-fragment needs keys 16h..16h+15 per lane; S^T holds {8h..+7,16+8h..+7}
// -> one packed-f16 shfl_xor(16) exchange per q-tile.
// Softmax in base-2: log2(e) folded into the Q projection scale.
// Causal mask only on the diagonal block.
// ---------------------------------------------------------------------------
union H8 { v8h h; int i[4]; };

template <bool MASK>
__device__ __forceinline__ void attn_block(
    const f16* __restrict__ K, const f16* __restrict__ V,
    const v16h (&bq)[2][2], v8f (&acc)[4][2], float (&mst)[2],
    float (&lst)[2], int kb, int q0, int mn, int half) {
  v16h ak[2][2];
#pragma unroll
  for (int ki = 0; ki < 2; ++ki)
#pragma unroll
    for (int ds = 0; ds < 2; ++ds)
      ak[ki][ds] = frag_a(K + (long)(kb + ki * 16 + mn) * DQ + ds * 32, half);

  v8f st[2][2] = {};  // [ki][qi]: lane = q (mn), VGPR j = key ki*16+8*half+j
#pragma unroll
  for (int ki = 0; ki < 2; ++ki)
#pragma unroll
    for (int qi = 0; qi < 2; ++qi) {
      st[ki][qi] = wmma_f16(ak[ki][0], bq[qi][0], st[ki][qi]);
      st[ki][qi] = wmma_f16(ak[ki][1], bq[qi][1], st[ki][qi]);
    }

  if (MASK) {
#pragma unroll
    for (int ki = 0; ki < 2; ++ki)
#pragma unroll
      for (int qi = 0; qi < 2; ++qi)
#pragma unroll
        for (int j = 0; j < 8; ++j) {
          const int key = kb + ki * 16 + 8 * half + j;
          const int q = q0 + qi * 16 + mn;
          if (key > q) st[ki][qi][j] = -1.0e30f;
        }
  }

  v16h bp[2];  // P^T as B fragments
#pragma unroll
  for (int qi = 0; qi < 2; ++qi) {
    float rm = st[0][qi][0];
#pragma unroll
    for (int j = 1; j < 8; ++j) rm = fmaxf(rm, st[0][qi][j]);
#pragma unroll
    for (int j = 0; j < 8; ++j) rm = fmaxf(rm, st[1][qi][j]);
    rm = fmaxf(rm, __shfl_xor(rm, 16, 32));
    const float mnew = fmaxf(mst[qi], rm);
    const float corr = __builtin_amdgcn_exp2f(mst[qi] - mnew);
    mst[qi] = mnew;
    float rs = 0.0f;
#pragma unroll
    for (int ki = 0; ki < 2; ++ki)
#pragma unroll
      for (int j = 0; j < 8; ++j) {
        const float e = __builtin_amdgcn_exp2f(st[ki][qi][j] - mnew);
        st[ki][qi][j] = e;
        rs += e;
      }
    rs += __shfl_xor(rs, 16, 32);
    lst[qi] = lst[qi] * corr + rs;

    // in-lane rescale of ctx^T accumulator (lane = q)
#pragma unroll
    for (int di = 0; di < 4; ++di)
#pragma unroll
      for (int j = 0; j < 8; ++j) acc[di][qi][j] *= corr;

    // Build P^T B-fragment: own half-keys + partner's via one xor-16 exchange.
    H8 own0, own1, send, recv;
#pragma unroll
    for (int j = 0; j < 8; ++j) {
      own0.h[j] = (f16)st[0][qi][j];  // keys 8*half + j
      own1.h[j] = (f16)st[1][qi][j];  // keys 16 + 8*half + j
    }
    send = half ? own0 : own1;
#pragma unroll
    for (int t = 0; t < 4; ++t) recv.i[t] = __shfl_xor(send.i[t], 16, 32);
#pragma unroll
    for (int j = 0; j < 8; ++j) {
      // elems 0..7 = keys 16*half .. +7 ; elems 8..15 = keys 16*half+8 .. +15
      bp[qi][j]     = half ? recv.h[j] : own0.h[j];
      bp[qi][8 + j] = half ? own1.h[j] : recv.h[j];
    }
  }

  v16h av[4];  // V^T tiles as A fragments (Vt is (D,T): key-contiguous)
#pragma unroll
  for (int di = 0; di < 4; ++di)
    av[di] = frag_a(V + (long)(di * 16 + mn) * TQ + kb, half);
#pragma unroll
  for (int di = 0; di < 4; ++di)
#pragma unroll
    for (int qi = 0; qi < 2; ++qi)
      acc[di][qi] = wmma_f16(av[di], bp[qi], acc[di][qi]);
}

__global__ __launch_bounds__(256)
void k_attn(const f16* __restrict__ Qh, const f16* __restrict__ Kh,
            const f16* __restrict__ Vt, f16* __restrict__ ctx) {
  const int lane = threadIdx.x & 31;
  const int wave = threadIdx.x >> 5;
  const int mn   = lane & 15;
  const int half = lane >> 4;
  const int b = blockIdx.z, h = blockIdx.y;
  const int q0 = blockIdx.x * 256 + wave * 32;

  const f16* Q = Qh + (long)(b * HQ + h) * TQ * DQ;  // (T,D), pre-scaled
  const f16* K = Kh + (long)(b * HQ + h) * TQ * DQ;  // (T,D)
  const f16* V = Vt + (long)(b * HQ + h) * DQ * TQ;  // (D,T)

  // Q as B fragments (lane = q col), held for the whole kernel.
  v16h bq[2][2];
#pragma unroll
  for (int qi = 0; qi < 2; ++qi)
#pragma unroll
    for (int ds = 0; ds < 2; ++ds)
      bq[qi][ds] = frag_b(Q + (long)(q0 + qi * 16 + mn) * DQ + ds * 32, half);

  float mst[2] = {-1.0e30f, -1.0e30f};
  float lst[2] = {0.0f, 0.0f};
  v8f acc[4][2] = {};  // ctx^T: [di][qi], lane = q, VGPR j = d 8*half+j

  for (int kb = 0; kb < q0; kb += 32)  // strictly-lower blocks: no masking
    attn_block<false>(K, V, bq, acc, mst, lst, kb, q0, mn, half);
  attn_block<true>(K, V, bq, acc, mst, lst, q0, q0, mn, half);  // diagonal

  // In-lane normalize; d runs along VGPR j -> contiguous 16B stores.
#pragma unroll
  for (int qi = 0; qi < 2; ++qi) {
    const float inv = __builtin_amdgcn_rcpf(lst[qi]);
    const int q = q0 + qi * 16 + mn;
#pragma unroll
    for (int di = 0; di < 4; ++di) {
      v8h o;
#pragma unroll
      for (int j = 0; j < 8; ++j) o[j] = (f16)(acc[di][qi][j] * inv);
      *(v8h*)(ctx + ((long)b * TQ + q) * CQ + h * DQ + di * 16 + 8 * half) = o;
    }
  }
}

// ---------------------------------------------------------------------------
// Host-side launch
// ---------------------------------------------------------------------------
extern "C" void kernel_launch(void* const* d_in, const int* in_sizes, int n_in,
                              void* d_out, int out_size, void* d_ws,
                              size_t ws_size, hipStream_t stream) {
  (void)in_sizes; (void)n_in; (void)out_size; (void)ws_size;
  const float* x  = (const float*)d_in[0];
  // d_in[1] = tril mask -- causal structure computed analytically
  const float* Wq = (const float*)d_in[2];
  const float* Wk = (const float*)d_in[3];
  const float* Wv = (const float*)d_in[4];
  const float* Wo = (const float*)d_in[5];
  const float* bo = (const float*)d_in[6];
  float* out = (float*)d_out;

  const long MB = 1024L * 1024L;
  char* ws = (char*)d_ws;
  f16* xh  = (f16*)(ws);            // 16 MB (B*T,C) f16; reused as ctx later
  f16* WqT = (f16*)(ws + 16 * MB);  // 2 MB each, N-major
  f16* WkT = (f16*)(ws + 18 * MB);
  f16* WvT = (f16*)(ws + 20 * MB);
  f16* WoT = (f16*)(ws + 22 * MB);
  f16* Qh  = (f16*)(ws + 24 * MB);  // 16 MB (B,H,T,D)
  f16* Kh  = (f16*)(ws + 40 * MB);  // 16 MB (B,H,T,D)
  f16* Vt  = (f16*)(ws + 56 * MB);  // 16 MB (B,H,D,T); total 72 MB
  f16* ctx = xh;                    // xh dead after projections

  k_cvt_f16<<<2048, 256, 0, stream>>>(x, xh, MTOT * CQ);
  const int twb = (CQ * CQ + 255) / 256;
  k_transpose_w<<<twb, 256, 0, stream>>>(Wq, WqT);
  k_transpose_w<<<twb, 256, 0, stream>>>(Wk, WkT);
  k_transpose_w<<<twb, 256, 0, stream>>>(Wv, WvT);
  k_transpose_w<<<twb, 256, 0, stream>>>(Wo, WoT);

  dim3 gg((int)(MTOT / 128), CQ / 128);
  // Q pre-scaled by log2(e)/sqrt(D): softmax runs in base-2 (v_exp_f32 native)
  k_gemm<0><<<gg, 256, 0, stream>>>(xh, WqT, Qh, nullptr, nullptr,
                                    0.125f * 1.44269504088896340736f);
  k_gemm<0><<<gg, 256, 0, stream>>>(xh, WkT, Kh, nullptr, nullptr, 1.0f);
  k_gemm<1><<<gg, 256, 0, stream>>>(xh, WvT, Vt, nullptr, nullptr, 1.0f);

  dim3 ga(TQ / 256, HQ, BQ);
  k_attn<<<ga, 256, 0, stream>>>(Qh, Kh, Vt, ctx);

  k_gemm<2><<<gg, 256, 0, stream>>>(ctx, WoT, nullptr, out, bo, 1.0f);
}
